// ECGraphNet_16655883174000
// MI455X (gfx1250) — matrix operand order repaired and checked by reading, not compiled, not run on hardware
//
#include <hip/hip_runtime.h>
#include <hip/hip_bf16.h>

// ---------------------------------------------------------------------------
// ECGraphNet forward for MI455X (gfx1250, wave32, WMMA).
// Shapes: B=2, C=256, H=W=64 (N=4096), NODE_NUM=32, K=5, 2C=512.
// Both 1x1 convs are bf16 WMMA GEMMs (f32 accumulate); graph aggregation is
// refactored to  invsig * (assign^T @ xf - anchor * sum(assign))  so it needs
// no [B,N,M,C] intermediate. Scratch usage ~77 MB (fits comfortably in the
// 192 MB L2 -> GEMMs are matrix-pipe bound, not HBM bound).
// ---------------------------------------------------------------------------

typedef __attribute__((ext_vector_type(16))) __bf16 v16bf;
typedef __attribute__((ext_vector_type(8)))  __bf16 v8bf;
typedef __attribute__((ext_vector_type(8)))  float  v8f;

union FragBF { v16bf v; v8bf h[2]; };

constexpr int CB = 2;        // batch
constexpr int CC = 256;      // channels
constexpr int CN = 4096;     // H*W
constexpr int CM = 32;       // NODE_NUM
constexpr int CK = 5;        // top-k
constexpr int C2 = 512;      // 2*C
constexpr int NK = CN * CK;  // 20480 columns for conv1

__device__ __forceinline__ unsigned short f32_to_bf16(float f) {
  unsigned int u = __builtin_bit_cast(unsigned int, f);
  u += 0x7fffu + ((u >> 16) & 1u);               // round-to-nearest-even
  return (unsigned short)(u >> 16);
}
__device__ __forceinline__ float bf16_to_f32(unsigned short h) {
  unsigned int u = ((unsigned int)h) << 16;
  return __builtin_bit_cast(float, u);
}

// ---------------------------------------------------------------------------
// x1 = sigmoid(edge) * x, converted to bf16 and stored transposed as
// [b][n][c] (K-contiguous per GEMM column) for the WMMA B operand.
// ---------------------------------------------------------------------------
__global__ __launch_bounds__(256)
void k_prep_x1(const float* __restrict__ x, const float* __restrict__ edge,
               unsigned short* __restrict__ x1bf) {
  const int tid = blockIdx.x * 256 + threadIdx.x;   // [b][n][c], c fastest
  const int c = tid & 255;
  const int n = (tid >> 8) & 4095;
  const int b = tid >> 20;
  const float e = edge[b * CN + n];
  const float s = 1.0f / (1.0f + __expf(-e));
  const float v = s * x[((size_t)(b * CC + c)) * CN + n];
  x1bf[tid] = f32_to_bf16(v);
}

__global__ __launch_bounds__(256)
void k_cvt_bf16(const float* __restrict__ src, unsigned short* __restrict__ dst,
                int n) {
  const int i = blockIdx.x * 256 + threadIdx.x;
  if (i < n) dst[i] = f32_to_bf16(src[i]);
}

// Store one 16x16 f32 accumulator per the ISA C/D layout.
template <bool OB>
__device__ __forceinline__ void store_acc(void* Out, size_t obase, int Ncols,
                                          int row0, int col, const v8f& acc) {
#pragma unroll
  for (int r = 0; r < 8; ++r) {
    const size_t idx = obase + (size_t)(row0 + r) * Ncols + col;
    if constexpr (OB)
      ((unsigned short*)Out)[idx] = f32_to_bf16(acc[r]);
    else
      ((float*)Out)[idx] = acc[r];
  }
}

// ---------------------------------------------------------------------------
// WMMA bf16 GEMM:  Out[z][o][j] = sum_k A[o][k] * Bt[z][j][k]
//   A : [Mr][KD] bf16 row-major (weights, shared across batch)
//   Bt: [batch][Ncols][KD] bf16 (columns K-contiguous -> contiguous b128 frags)
// Block = 256 threads = 8 waves. Block tile 16(M) x 512(N): each wave owns 4
// column tiles, so one LDS A fragment feeds 4 back-to-back WMMAs. The 16x32 A
// tile is double-buffered in LDS (threads 0..63 copy one b128 chunk each) ->
// one barrier per K-step. KD is a template constant so all strides fold.
// ---------------------------------------------------------------------------
template <int KD, bool OUT_BF16>
__global__ __launch_bounds__(256)
void k_wmma_gemm(const unsigned short* __restrict__ A,
                 const unsigned short* __restrict__ Bt,
                 void* __restrict__ Out, int Mr, int Ncols) {
  const int lane = threadIdx.x & 31;
  const int wave = threadIdx.x >> 5;
  const int rowBase = blockIdx.y << 4;
  const int colBase = (blockIdx.x << 9) + (wave << 6);

  const unsigned short* Bb = Bt + (size_t)blockIdx.z * (size_t)Ncols * KD;

  __shared__ unsigned short As[2][16 * 32];   // double-buffered A tile

  // ISA 16-bit A layout: lanes 0-15 row=lane, K {0..7,16..23};
  //                      lanes 16-31 row=lane-16, K {8..15,24..31}.
  const int rowA = lane & 15;
  const int aoff = (lane >= 16) ? 8 : 0;
  // ISA 16-bit B layout: lanes 0-15 col=lane, K {0..7,16..23}; lanes 16-31 K {8..15,24..31}.
  const int colB = colBase + (lane & 15);
  const int boff = (lane >= 16) ? 8 : 0;
  const unsigned short* bcol = Bb + (size_t)colB * KD + boff;

  // Cooperative A-tile loader: threads 0..63 move one b128 chunk each.
  const int ldr_row = threadIdx.x >> 2;         // 0..15
  const int ldr_ch = (threadIdx.x & 3) << 3;    // 0,8,16,24
  const unsigned short* aldr = A + (size_t)(rowBase + ldr_row) * KD + ldr_ch;

  if (threadIdx.x < 64)
    *(v8bf*)&As[0][ldr_row * 32 + ldr_ch] = *(const v8bf*)aldr;

  v8f acc0 = {}, acc1 = {}, acc2 = {}, acc3 = {};

  int cur = 0;
  for (int kt = 0; kt < KD; kt += 32, cur ^= 1) {
    __syncthreads();  // As[cur] ready for everyone; previous reads drained
    if (kt + 32 < KD && threadIdx.x < 64)
      *(v8bf*)&As[cur ^ 1][ldr_row * 32 + ldr_ch] =
          *(const v8bf*)(aldr + kt + 32);

    FragBF a;
    a.h[0] = *(const v8bf*)&As[cur][rowA * 32 + aoff];
    a.h[1] = *(const v8bf*)&As[cur][rowA * 32 + 16 + aoff];

    __builtin_prefetch(bcol + kt + 64, 0, 3);   // next-next K-step B fragment

    FragBF b0, b1, b2, b3;
    b0.h[0] = *(const v8bf*)(bcol + kt);
    b0.h[1] = *(const v8bf*)(bcol + kt + 16);
    b1.h[0] = *(const v8bf*)(bcol + 16 * KD + kt);
    b1.h[1] = *(const v8bf*)(bcol + 16 * KD + kt + 16);
    b2.h[0] = *(const v8bf*)(bcol + 32 * KD + kt);
    b2.h[1] = *(const v8bf*)(bcol + 32 * KD + kt + 16);
    b3.h[0] = *(const v8bf*)(bcol + 48 * KD + kt);
    b3.h[1] = *(const v8bf*)(bcol + 48 * KD + kt + 16);

    acc0 = __builtin_amdgcn_wmma_f32_16x16x32_bf16(false, a.v, false, b0.v,
                                                   (short)0, acc0, false, false);
    acc1 = __builtin_amdgcn_wmma_f32_16x16x32_bf16(false, a.v, false, b1.v,
                                                   (short)0, acc1, false, false);
    acc2 = __builtin_amdgcn_wmma_f32_16x16x32_bf16(false, a.v, false, b2.v,
                                                   (short)0, acc2, false, false);
    acc3 = __builtin_amdgcn_wmma_f32_16x16x32_bf16(false, a.v, false, b3.v,
                                                   (short)0, acc3, false, false);
  }

  // C/D layout: lanes 0-15 -> M = r, lanes 16-31 -> M = r+8; N = lane&15.
  const int row0 = rowBase + ((lane >= 16) ? 8 : 0);
  const int col = colBase + (lane & 15);
  const size_t obase = (size_t)blockIdx.z * (size_t)Mr * Ncols;
  store_acc<OUT_BF16>(Out, obase, Ncols, row0, col + 0, acc0);
  store_acc<OUT_BF16>(Out, obase, Ncols, row0, col + 16, acc1);
  store_acc<OUT_BF16>(Out, obase, Ncols, row0, col + 32, acc2);
  store_acc<OUT_BF16>(Out, obase, Ncols, row0, col + 48, acc3);
}

// ---------------------------------------------------------------------------
// Train-mode BN statistics (per channel over batch * columns).
// ---------------------------------------------------------------------------
__global__ __launch_bounds__(256)
void k_stats_f32(const float* __restrict__ y, int Ncol, float* __restrict__ mean,
                 float* __restrict__ var) {
  const int c = blockIdx.x, t = threadIdx.x;
  __shared__ float r1[256], r2[256];
  float s = 0.f, s2 = 0.f;
  for (int b = 0; b < CB; ++b) {
    const float* p = y + ((size_t)(b * CC + c)) * Ncol;
    for (int j = t; j < Ncol; j += 256) { float v = p[j]; s += v; s2 += v * v; }
  }
  r1[t] = s; r2[t] = s2; __syncthreads();
  for (int st = 128; st > 0; st >>= 1) {
    if (t < st) { r1[t] += r1[t + st]; r2[t] += r2[t + st]; }
    __syncthreads();
  }
  if (t == 0) {
    const float cnt = (float)(CB * Ncol);
    const float m = r1[0] / cnt;
    mean[c] = m;
    var[c] = fmaxf(r2[0] / cnt - m * m, 0.f);
  }
}

__global__ __launch_bounds__(256)
void k_stats_bf16(const unsigned short* __restrict__ y, int Ncol,
                  float* __restrict__ mean, float* __restrict__ var) {
  const int c = blockIdx.x, t = threadIdx.x;
  __shared__ float r1[256], r2[256];
  float s = 0.f, s2 = 0.f;
  for (int b = 0; b < CB; ++b) {
    const unsigned short* p = y + ((size_t)(b * CC + c)) * Ncol;
    for (int j = t; j < Ncol; j += 256) {
      float v = bf16_to_f32(p[j]); s += v; s2 += v * v;
    }
  }
  r1[t] = s; r2[t] = s2; __syncthreads();
  for (int st = 128; st > 0; st >>= 1) {
    if (t < st) { r1[t] += r1[t + st]; r2[t] += r2[t + st]; }
    __syncthreads();
  }
  if (t == 0) {
    const float cnt = (float)(CB * Ncol);
    const float m = r1[0] / cnt;
    mean[c] = m;
    var[c] = fmaxf(r2[0] / cnt - m * m, 0.f);
  }
}

// In-place BN + ReLU on y0 [B][C][N].
__global__ __launch_bounds__(256)
void k_bnrelu(float* __restrict__ y, const float* __restrict__ mean,
              const float* __restrict__ var, const float* __restrict__ g,
              const float* __restrict__ bb) {
  const int tid = blockIdx.x * 256 + threadIdx.x;
  const int c = (tid >> 12) & 255;
  const float sc = g[c] * rsqrtf(var[c] + 1e-5f);
  const float bi = bb[c] - mean[c] * sc;
  y[tid] = fmaxf(y[tid] * sc + bi, 0.f);
}

// invsig = 1/sigmoid(sigma_p) = 1 + exp(-sigma_p)
__global__ __launch_bounds__(256)
void k_invsig(const float* __restrict__ sp, float* __restrict__ invsig) {
  const int i = blockIdx.x * 256 + threadIdx.x;
  invsig[i] = 1.0f + __expf(-sp[i]);
}

// Per-pixel soft-assignment: logits[m] = -0.5*sum_c ((xf_c - a_mc)*invsig_mc)^2
__global__ __launch_bounds__(256)
void k_assign(const float* __restrict__ xf, const float* __restrict__ anchor,
              const float* __restrict__ invsig, float* __restrict__ assign) {
  const int b = blockIdx.x >> 12;
  const int n = blockIdx.x & 4095;
  const int t = threadIdx.x;  // channel
  __shared__ float red[256];
  __shared__ float logit[CM];
  const float xc = xf[((size_t)(b * CC + t)) * CN + n];
  for (int m = 0; m < CM; ++m) {
    const float r = (xc - anchor[m * CC + t]) * invsig[m * CC + t];
    red[t] = r * r;
    __syncthreads();
    for (int s = 128; s > 0; s >>= 1) {
      if (t < s) red[t] += red[t + s];
      __syncthreads();
    }
    if (t == 0) logit[m] = -0.5f * red[0];
    __syncthreads();
  }
  if (t == 0) {
    float mx = logit[0];
    for (int m = 1; m < CM; ++m) mx = fmaxf(mx, logit[m]);
    float s = 0.f;
    for (int m = 0; m < CM; ++m) { float e = __expf(logit[m] - mx); logit[m] = e; s += e; }
    const float inv = 1.f / s;
    for (int m = 0; m < CM; ++m) logit[m] *= inv;
  }
  __syncthreads();
  if (t < CM) assign[(size_t)blockIdx.x * CM + t] = logit[t];
}

__global__ __launch_bounds__(256)
void k_asum(const float* __restrict__ assign, float* __restrict__ asum) {
  const int b = blockIdx.x >> 5;
  const int m = blockIdx.x & 31;
  const int t = threadIdx.x;
  __shared__ float red[256];
  float s = 0.f;
  for (int n = t; n < CN; n += 256) s += assign[((size_t)b * CN + n) * CM + m];
  red[t] = s; __syncthreads();
  for (int st = 128; st > 0; st >>= 1) {
    if (t < st) red[t] += red[t + st];
    __syncthreads();
  }
  if (t == 0) asum[blockIdx.x] = red[0];
}

// nodes[b][m][c] = invsig*(sum_n a[n,m]*xf[c,n] - anchor*asum) / (asum+eps)
__global__ __launch_bounds__(256)
void k_nodes(const float* __restrict__ assign, const float* __restrict__ xf,
             const float* __restrict__ invsig, const float* __restrict__ anchor,
             const float* __restrict__ asum, float* __restrict__ nodes) {
  const int tid = blockIdx.x * 256 + threadIdx.x;
  const int c = tid & 255;
  const int m = (tid >> 8) & 31;
  const int b = tid >> 13;
  const float* ap = assign + (size_t)b * CN * CM + m;
  const float* xp = xf + ((size_t)(b * CC + c)) * CN;
  float acc = 0.f;
  for (int n = 0; n < CN; ++n) acc += ap[(size_t)n * CM] * xp[n];
  const float s = asum[b * CM + m];
  float node = invsig[m * CC + c] * (acc - anchor[m * CC + c] * s);
  nodes[tid] = node / (s + 1e-9f);
}

// L2-normalize each node (over C), in place.
__global__ __launch_bounds__(256)
void k_norm_nodes(float* __restrict__ nodes) {
  const int t = threadIdx.x;
  __shared__ float red[256];
  __shared__ float nrm;
  const float v = nodes[(size_t)blockIdx.x * CC + t];
  red[t] = v * v; __syncthreads();
  for (int s = 128; s > 0; s >>= 1) {
    if (t < s) red[t] += red[t + s];
    __syncthreads();
  }
  if (t == 0) nrm = fmaxf(sqrtf(red[0]), 1e-12f);
  __syncthreads();
  nodes[(size_t)blockIdx.x * CC + t] = v / nrm;
}

__global__ __launch_bounds__(256)
void k_flatnorm(const float* __restrict__ nodes, float* __restrict__ fl) {
  const int b = blockIdx.x, t = threadIdx.x;
  __shared__ float red[256];
  float s = 0.f;
  for (int i = t; i < CM * CC; i += 256) {
    const float v = nodes[(size_t)b * CM * CC + i];
    s += v * v;
  }
  red[t] = s; __syncthreads();
  for (int st = 128; st > 0; st >>= 1) {
    if (t < st) red[t] += red[t + st];
    __syncthreads();
  }
  if (t == 0) fl[b] = fmaxf(sqrtf(red[0]), 1e-12f);
}

// Faithful torch .view(B,C,M).transpose: nt[b][m][c] = flatbuf[b][c*32+m]/fl[b]
__global__ __launch_bounds__(256)
void k_permute(const float* __restrict__ nodes, const float* __restrict__ fl,
               float* __restrict__ nt) {
  const int tid = blockIdx.x * 256 + threadIdx.x;  // (b, m, c), c fastest
  const int c = tid & 255;
  const int m = (tid >> 8) & 31;
  const int b = tid >> 13;
  nt[tid] = nodes[(size_t)b * CM * CC + c * CM + m] / fl[b];
}

// Per-pixel distances to 32 nodes, rank-select the 5 smallest (tie order
// matches jax top_k: lower index wins), then fuse gather+concat into the
// bf16 B operand of conv1:  cat[b,n,k] = [nt[sel_k]-x , x].
__global__ __launch_bounds__(256)
void k_topk_cat(const float* __restrict__ x, const float* __restrict__ nt,
                unsigned short* __restrict__ catbf) {
  const int b = blockIdx.x >> 12;
  const int n = blockIdx.x & 4095;
  const int t = threadIdx.x;  // channel
  __shared__ float red[256];
  __shared__ float dist[CM];
  __shared__ int sel[CK];
  const float xc = x[((size_t)(b * CC + t)) * CN + n];
  for (int m = 0; m < CM; ++m) {
    const float d = xc - nt[(size_t)(b * CM + m) * CC + t];
    red[t] = d * d;
    __syncthreads();
    for (int s = 128; s > 0; s >>= 1) {
      if (t < s) red[t] += red[t + s];
      __syncthreads();
    }
    if (t == 0) dist[m] = red[0];
    __syncthreads();
  }
  if (t < CM) {
    const float dm = dist[t];
    int rank = 0;
    for (int j = 0; j < CM; ++j) {
      const float dj = dist[j];
      rank += (dj < dm) || (dj == dm && j < t);
    }
    if (rank < CK) sel[rank] = t;
  }
  __syncthreads();
  const size_t base = (size_t)blockIdx.x * CK * C2;
  for (int k = 0; k < CK; ++k) {
    const float g = nt[(size_t)(b * CM + sel[k]) * CC + t];
    catbf[base + (size_t)k * C2 + t] = f32_to_bf16(g - xc);
    catbf[base + (size_t)k * C2 + CC + t] = f32_to_bf16(xc);
  }
}

// out = x + max_k relu(bn1(y1[b][c][n*5+k]))
__global__ __launch_bounds__(256)
void k_final(const float* __restrict__ x, const unsigned short* __restrict__ y1,
             const float* __restrict__ mean, const float* __restrict__ var,
             const float* __restrict__ g, const float* __restrict__ bb,
             float* __restrict__ out) {
  const int tid = blockIdx.x * 256 + threadIdx.x;
  const int n = tid & 4095;
  const int c = (tid >> 12) & 255;
  const int b = tid >> 20;
  const float sc = g[c] * rsqrtf(var[c] + 1e-5f);
  const float bi = bb[c] - mean[c] * sc;
  const unsigned short* yp = y1 + ((size_t)(b * CC + c)) * NK + (size_t)n * CK;
  float best = -1e30f;
#pragma unroll
  for (int k = 0; k < CK; ++k) best = fmaxf(best, bf16_to_f32(yp[k]) * sc + bi);
  out[tid] = x[tid] + fmaxf(best, 0.f);
}

// ---------------------------------------------------------------------------
extern "C" void kernel_launch(void* const* d_in, const int* in_sizes, int n_in,
                              void* d_out, int out_size, void* d_ws,
                              size_t ws_size, hipStream_t stream) {
  (void)in_sizes; (void)n_in; (void)out_size; (void)ws_size;
  const float* x      = (const float*)d_in[0];
  const float* edge   = (const float*)d_in[1];
  const float* w0     = (const float*)d_in[2];
  const float* g0     = (const float*)d_in[3];
  const float* b0     = (const float*)d_in[4];
  const float* anchor = (const float*)d_in[5];
  const float* sigp   = (const float*)d_in[6];
  const float* w1     = (const float*)d_in[7];
  const float* g1     = (const float*)d_in[8];
  const float* b1     = (const float*)d_in[9];
  float* out = (float*)d_out;

  // Workspace carve-up (~77 MB total, 256B aligned slices).
  char* p = (char*)d_ws;
  auto take = [&](size_t bytes) -> char* {
    char* r = p;
    p += (bytes + 255) & ~(size_t)255;
    return r;
  };
  unsigned short* x1bf  = (unsigned short*)take(2ull * CB * CN * CC);
  unsigned short* w0bf  = (unsigned short*)take(2ull * CC * CC);
  float* y0             = (float*)take(4ull * CB * CC * CN);
  float* mean0          = (float*)take(4ull * CC);
  float* var0           = (float*)take(4ull * CC);
  float* invsig         = (float*)take(4ull * CM * CC);
  float* assign         = (float*)take(4ull * CB * CN * CM);
  float* asum           = (float*)take(4ull * CB * CM);
  float* nodesb         = (float*)take(4ull * CB * CM * CC);
  float* flnorm         = (float*)take(4ull * CB);
  float* nt             = (float*)take(4ull * CB * CM * CC);
  unsigned short* catbf = (unsigned short*)take(2ull * CB * NK * C2);
  unsigned short* w1bf  = (unsigned short*)take(2ull * CC * C2);
  unsigned short* y1bf  = (unsigned short*)take(2ull * CB * CC * NK);
  float* mean1          = (float*)take(4ull * CC);
  float* var1           = (float*)take(4ull * CC);

  // 1. edge-gating + bf16 transpose; weight conversions
  k_prep_x1<<<(CB * CN * CC) / 256, 256, 0, stream>>>(x, edge, x1bf);
  k_cvt_bf16<<<(CC * CC) / 256, 256, 0, stream>>>(w0, w0bf, CC * CC);
  k_cvt_bf16<<<(CC * C2) / 256, 256, 0, stream>>>(w1, w1bf, CC * C2);

  // 2. conv0 GEMM (WMMA bf16): y0[b][o][n]
  dim3 gg0(CN / 512, CC / 16, CB);
  k_wmma_gemm<CC, false><<<gg0, 256, 0, stream>>>(w0bf, x1bf, y0, CC, CN);

  // 3. BN0 (train-mode) + ReLU, in place
  k_stats_f32<<<CC, 256, 0, stream>>>(y0, CN, mean0, var0);
  k_bnrelu<<<(CB * CC * CN) / 256, 256, 0, stream>>>(y0, mean0, var0, g0, b0);

  // 4. GraphNet
  k_invsig<<<(CM * CC) / 256, 256, 0, stream>>>(sigp, invsig);
  k_assign<<<CB * CN, 256, 0, stream>>>(y0, anchor, invsig, assign);
  k_asum<<<CB * CM, 256, 0, stream>>>(assign, asum);
  k_nodes<<<(CB * CM * CC) / 256, 256, 0, stream>>>(assign, y0, invsig, anchor,
                                                    asum, nodesb);
  k_norm_nodes<<<CB * CM, 256, 0, stream>>>(nodesb);
  k_flatnorm<<<CB, 256, 0, stream>>>(nodesb, flnorm);
  k_permute<<<(CB * CM * CC) / 256, 256, 0, stream>>>(nodesb, flnorm, nt);

  // 5. distances + top-5 + fused gather/concat into bf16 cat
  k_topk_cat<<<CB * CN, 256, 0, stream>>>(x, nt, catbf);

  // 6. conv1 GEMM (WMMA bf16): y1[b][o][n*5+k]
  dim3 gg1(NK / 512, CC / 16, CB);
  k_wmma_gemm<C2, true><<<gg1, 256, 0, stream>>>(w1bf, catbf, y1bf, CC, NK);

  // 7. BN1 stats, then BN+ReLU+max-over-K+residual
  k_stats_bf16<<<CC, 256, 0, stream>>>(y1bf, NK, mean1, var1);
  k_final<<<(CB * CC * CN) / 256, 256, 0, stream>>>(x, y1bf, mean1, var1, g1,
                                                    b1, out);
}
